// MultiHeadAttention_57569741635947
// MI455X (gfx1250) — matrix operand order repaired and checked by reading, not compiled, not run on hardware
//
#include <hip/hip_runtime.h>
#include <hip/hip_bf16.h>

typedef __bf16 bf16;
typedef __attribute__((ext_vector_type(16))) __bf16 v16bf;
typedef __attribute__((ext_vector_type(8)))  __bf16 v8bf;
typedef __attribute__((ext_vector_type(8)))  float  v8f;
typedef __attribute__((ext_vector_type(4)))  float  v4f;

#define BATCH 4
#define SEQL  2048
#define HEADS 16
#define DKH   64
#define DM    1024

// ---------------------------------------------------------------------------
// helpers
// ---------------------------------------------------------------------------
__device__ __forceinline__ v16bf cat16(v8bf lo, v8bf hi) {
  v16bf f;
#pragma unroll
  for (int i = 0; i < 8; ++i) { f[i] = lo[i]; f[i + 8] = hi[i]; }
  return f;
}

// A-matrix 16x32 bf16 fragment (lane m = L&15; K groups {kb..kb+7, 16+kb..23+kb}, kb = (L&16)?8:0)
__device__ __forceinline__ v16bf load_a_bf16(const bf16* base, int ld, int row0, int k0, int lane) {
  int m  = lane & 15;
  int kb = (lane & 16) ? 8 : 0;
  const bf16* p = base + (size_t)(row0 + m) * ld + k0 + kb;
  v8bf lo = *(const v8bf*)p;
  v8bf hi = *(const v8bf*)(p + 16);
  return cat16(lo, hi);
}

// Same A fragment but converting from fp32 source on the fly.
__device__ __forceinline__ v16bf load_a_f32(const float* base, int ld, int row0, int k0, int lane) {
  int m  = lane & 15;
  int kb = (lane & 16) ? 8 : 0;
  const float* p = base + (size_t)(row0 + m) * ld + k0 + kb;
  v4f a0 = *(const v4f*)(p);
  v4f a1 = *(const v4f*)(p + 4);
  v4f b0 = *(const v4f*)(p + 16);
  v4f b1 = *(const v4f*)(p + 20);
  v16bf f;
#pragma unroll
  for (int i = 0; i < 4; ++i) {
    f[i]      = (bf16)a0[i];
    f[i + 4]  = (bf16)a1[i];
    f[i + 8]  = (bf16)b0[i];
    f[i + 12] = (bf16)b1[i];
  }
  return f;
}

// B-matrix 32x16 bf16 fragment. Logical B(k,n) = M[n*ld + k] (row n of M is column n of B).
// Lane n = L&15 holds K groups {kh..kh+7, 16+kh..23+kh}, kh = (L&16)?8:0.
__device__ __forceinline__ v16bf load_b_bf16(const bf16* base, int ld, int n0, int k0, int lane) {
  int n  = lane & 15;
  int kh = (lane & 16) ? 8 : 0;
  const bf16* p = base + (size_t)(n0 + n) * ld + k0 + kh;
  v8bf lo = *(const v8bf*)p;
  v8bf hi = *(const v8bf*)(p + 16);
  return cat16(lo, hi);
}

__device__ __forceinline__ v8f wmma_bf16(v16bf a, v16bf b, v8f c) {
  return __builtin_amdgcn_wmma_f32_16x16x32_bf16(false, a, false, b, (short)0, c, false, false);
}

// ---------------------------------------------------------------------------
// elementwise: fp32 -> bf16
// ---------------------------------------------------------------------------
__global__ void cvt_f32_bf16(const float* __restrict__ in, bf16* __restrict__ out, int n) {
  int i = blockIdx.x * blockDim.x + threadIdx.x;
  if (i < n) out[i] = (bf16)in[i];
}

// ---------------------------------------------------------------------------
// V (B,H,S,dk) -> Vt (B,H,dk,S)
// ---------------------------------------------------------------------------
__global__ void vtrans(const bf16* __restrict__ v, bf16* __restrict__ vt) {
  int i = blockIdx.x * blockDim.x + threadIdx.x;
  const int per_bh = SEQL * DKH;  // 131072
  int bh = i >> 17;
  int r  = i & (per_bh - 1);
  int s  = r >> 6;
  int d  = r & 63;
  vt[(size_t)bh * per_bh + (size_t)d * SEQL + s] = v[i];
}

// ---------------------------------------------------------------------------
// fused QKV projection: O = X @ W^T + b, stored as (B,H,S,dk) bf16
// grid (M/128, N/128, 3), block 128 (4 waves, 2x2 of 64x64 wave tiles)
// ---------------------------------------------------------------------------
__global__ __launch_bounds__(128) void proj_qkv(
    const float* __restrict__ Xq, const float* __restrict__ Xk, const float* __restrict__ Xv,
    const bf16* __restrict__ Wall,
    const float* __restrict__ biasq, const float* __restrict__ biask, const float* __restrict__ biasv,
    bf16* __restrict__ qo, bf16* __restrict__ ko, bf16* __restrict__ vo) {
  int z = blockIdx.z;
  const float* X    = (z == 0) ? Xq : (z == 1) ? Xk : Xv;
  const bf16*  W    = Wall + (size_t)z * DM * DM;
  const float* bias = (z == 0) ? biasq : (z == 1) ? biask : biasv;
  bf16*        O    = (z == 0) ? qo : (z == 1) ? ko : vo;

  int wave = threadIdx.x >> 5;
  int lane = threadIdx.x & 31;
  int row0 = blockIdx.x * 128 + (wave >> 1) * 64;
  int n00  = blockIdx.y * 128 + (wave & 1) * 64;

  v8f acc[4][4];
#pragma unroll
  for (int i = 0; i < 4; ++i)
#pragma unroll
    for (int j = 0; j < 4; ++j) acc[i][j] = (v8f)0.0f;

  for (int k0 = 0; k0 < DM; k0 += 32) {
    v16bf a[4], bb[4];
#pragma unroll
    for (int i = 0; i < 4; ++i) a[i] = load_a_f32(X, DM, row0 + i * 16, k0, lane);
#pragma unroll
    for (int j = 0; j < 4; ++j) bb[j] = load_b_bf16(W, DM, n00 + j * 16, k0, lane);
#pragma unroll
    for (int i = 0; i < 4; ++i)
#pragma unroll
      for (int j = 0; j < 4; ++j) acc[i][j] = wmma_bf16(a[i], bb[j], acc[i][j]);
  }

  int mo = (lane & 16) ? 8 : 0;
  int nc = lane & 15;
#pragma unroll
  for (int j = 0; j < 4; ++j) {
    int n = n00 + j * 16 + nc;
    float bval = bias[n];
    int h = n >> 6, d = n & 63;
#pragma unroll
    for (int i = 0; i < 4; ++i) {
#pragma unroll
      for (int r = 0; r < 8; ++r) {
        int t = row0 + i * 16 + r + mo;
        int b = t >> 11, s = t & (SEQL - 1);
        size_t idx = (((size_t)(b * HEADS + h)) * SEQL + s) * DKH + d;
        O[idx] = (bf16)(acc[i][j][r] + bval);
      }
    }
  }
}

// ---------------------------------------------------------------------------
// flash attention, causal. 1 wave per 16-row Q tile, KV blocks of 64.
// grid 2048, block 128 (4 waves, independent tiles)
// ---------------------------------------------------------------------------
__global__ __launch_bounds__(128) void attn_kernel(
    const bf16* __restrict__ qb, const bf16* __restrict__ kmat,
    const bf16* __restrict__ vtb, bf16* __restrict__ attn) {
  __shared__ __attribute__((aligned(16))) bf16 Pt[4][16 * 72];

  int wave = threadIdx.x >> 5;
  int lane = threadIdx.x & 31;
  int tile = blockIdx.x * 4 + wave;
  int bh = tile >> 7;          // 128 q-tiles per (b,h)
  int q0 = (tile & 127) << 4;

  const bf16* Q  = qb   + (size_t)bh * SEQL * DKH;
  const bf16* K  = kmat + (size_t)bh * SEQL * DKH;
  const bf16* Vt = vtb  + (size_t)bh * DKH * SEQL;
  bf16* Pw = &Pt[wave][0];

  v16bf aq0 = load_a_bf16(Q, DKH, q0, 0, lane);
  v16bf aq1 = load_a_bf16(Q, DKH, q0, 32, lane);

  v8f acc[4];
#pragma unroll
  for (int j = 0; j < 4; ++j) acc[j] = (v8f)0.0f;
  float rmax[8], rsum[8];
#pragma unroll
  for (int r = 0; r < 8; ++r) { rmax[r] = -__builtin_inff(); rsum[r] = 0.0f; }

  int mo = (lane & 16) ? 8 : 0;
  int nc = lane & 15;
  const float scale = 0.125f;  // 1/sqrt(64)

  for (int j0 = 0; j0 <= q0 + 15; j0 += 64) {
    // ---- scores S = Q K^T for 16x64 block
    v8f sc[4];
#pragma unroll
    for (int f = 0; f < 4; ++f) sc[f] = (v8f)0.0f;
#pragma unroll
    for (int f = 0; f < 4; ++f) {
      v16bf bk0 = load_b_bf16(K, DKH, j0 + f * 16, 0, lane);
      v16bf bk1 = load_b_bf16(K, DKH, j0 + f * 16, 32, lane);
      sc[f] = wmma_bf16(aq0, bk0, sc[f]);
      sc[f] = wmma_bf16(aq1, bk1, sc[f]);
    }
    bool need_mask = (j0 + 63) > q0;
#pragma unroll
    for (int f = 0; f < 4; ++f)
#pragma unroll
      for (int r = 0; r < 8; ++r) {
        float s = sc[f][r] * scale;
        if (need_mask && (j0 + f * 16 + nc) > (q0 + r + mo)) s = -__builtin_inff();
        sc[f][r] = s;
      }
    // ---- block row max (rows live per VGPR r within each 16-lane half)
    float bm[8];
#pragma unroll
    for (int r = 0; r < 8; ++r)
      bm[r] = fmaxf(fmaxf(sc[0][r], sc[1][r]), fmaxf(sc[2][r], sc[3][r]));
#pragma unroll
    for (int msk = 1; msk <= 8; msk <<= 1)
#pragma unroll
      for (int r = 0; r < 8; ++r) bm[r] = fmaxf(bm[r], __shfl_xor(bm[r], msk, 32));

    float alpha[8];
#pragma unroll
    for (int r = 0; r < 8; ++r) {
      float nm = fmaxf(rmax[r], bm[r]);
      alpha[r] = __expf(rmax[r] - nm);
      rmax[r] = nm;
    }
    // ---- P = exp(S - max): stash bf16 P tile to LDS (C-layout -> memory)
    float bs[8];
#pragma unroll
    for (int r = 0; r < 8; ++r) bs[r] = 0.0f;
#pragma unroll
    for (int f = 0; f < 4; ++f)
#pragma unroll
      for (int r = 0; r < 8; ++r) {
        float p = __expf(sc[f][r] - rmax[r]);
        bs[r] += p;
        Pw[(r + mo) * 72 + f * 16 + nc] = (bf16)p;
      }
#pragma unroll
    for (int msk = 1; msk <= 8; msk <<= 1)
#pragma unroll
      for (int r = 0; r < 8; ++r) bs[r] += __shfl_xor(bs[r], msk, 32);
#pragma unroll
    for (int r = 0; r < 8; ++r) rsum[r] = rsum[r] * alpha[r] + bs[r];
#pragma unroll
    for (int j = 0; j < 4; ++j)
#pragma unroll
      for (int r = 0; r < 8; ++r) acc[j][r] *= alpha[r];

    // make the ds stores visible before re-reading as A fragments (same wave)
    asm volatile("s_wait_dscnt 0" ::: "memory");

    // ---- O += P V  (P re-read from LDS in A-fragment layout)
    v16bf pa0 = load_a_bf16(Pw, 72, 0, 0, lane);
    v16bf pa1 = load_a_bf16(Pw, 72, 0, 32, lane);
#pragma unroll
    for (int j = 0; j < 4; ++j) {
      v16bf bv0 = load_b_bf16(Vt, SEQL, j * 16, j0, lane);
      v16bf bv1 = load_b_bf16(Vt, SEQL, j * 16, j0 + 32, lane);
      acc[j] = wmma_bf16(pa0, bv0, acc[j]);
      acc[j] = wmma_bf16(pa1, bv1, acc[j]);
    }
  }

  // ---- normalize and store attn (B,S,D) bf16
  int b = bh >> 4, h = bh & 15;
#pragma unroll
  for (int j = 0; j < 4; ++j)
#pragma unroll
    for (int r = 0; r < 8; ++r) {
      int row = q0 + r + mo;
      float v = acc[j][r] / rsum[r];
      attn[((size_t)(b * SEQL + row)) * DM + h * DKH + j * 16 + nc] = (bf16)v;
    }
}

// ---------------------------------------------------------------------------
// output projection: out(f32) = attn(bf16) @ w_o^T + b_o
// ---------------------------------------------------------------------------
__global__ __launch_bounds__(128) void out_proj(
    const bf16* __restrict__ A, const bf16* __restrict__ W,
    const float* __restrict__ bias, float* __restrict__ out) {
  int wave = threadIdx.x >> 5;
  int lane = threadIdx.x & 31;
  int row0 = blockIdx.x * 128 + (wave >> 1) * 64;
  int n00  = blockIdx.y * 128 + (wave & 1) * 64;

  v8f acc[4][4];
#pragma unroll
  for (int i = 0; i < 4; ++i)
#pragma unroll
    for (int j = 0; j < 4; ++j) acc[i][j] = (v8f)0.0f;

  for (int k0 = 0; k0 < DM; k0 += 32) {
    v16bf a[4], bb[4];
#pragma unroll
    for (int i = 0; i < 4; ++i) a[i] = load_a_bf16(A, DM, row0 + i * 16, k0, lane);
#pragma unroll
    for (int j = 0; j < 4; ++j) bb[j] = load_b_bf16(W, DM, n00 + j * 16, k0, lane);
#pragma unroll
    for (int i = 0; i < 4; ++i)
#pragma unroll
      for (int j = 0; j < 4; ++j) acc[i][j] = wmma_bf16(a[i], bb[j], acc[i][j]);
  }

  int mo = (lane & 16) ? 8 : 0;
  int nc = lane & 15;
#pragma unroll
  for (int j = 0; j < 4; ++j) {
    int n = n00 + j * 16 + nc;
    float bval = bias[n];
#pragma unroll
    for (int i = 0; i < 4; ++i)
#pragma unroll
      for (int r = 0; r < 8; ++r) {
        int t = row0 + i * 16 + r + mo;
        out[(size_t)t * DM + n] = acc[i][j][r] + bval;
      }
  }
}

// ---------------------------------------------------------------------------
extern "C" void kernel_launch(void* const* d_in, const int* in_sizes, int n_in,
                              void* d_out, int out_size, void* d_ws, size_t ws_size,
                              hipStream_t stream) {
  const float* query = (const float*)d_in[0];
  const float* key   = (const float*)d_in[1];
  const float* value = (const float*)d_in[2];
  // d_in[3] = mask (causal, known statically) -- ignored
  const float* w_q = (const float*)d_in[4];
  const float* b_q = (const float*)d_in[5];
  const float* w_k = (const float*)d_in[6];
  const float* b_k = (const float*)d_in[7];
  const float* w_v = (const float*)d_in[8];
  const float* b_v = (const float*)d_in[9];
  const float* w_o = (const float*)d_in[10];
  const float* b_o = (const float*)d_in[11];

  const size_t EL = (size_t)BATCH * HEADS * SEQL * DKH;  // 8,388,608
  bf16* qbf  = (bf16*)d_ws;
  bf16* kbf  = qbf  + EL;
  bf16* vbf  = kbf  + EL;
  bf16* vtb  = vbf  + EL;
  bf16* attn = vtb  + EL;
  bf16* wbf  = attn + EL;  // w_q,w_k,w_v,w_o bf16, 1M elements each

  const int WEL = DM * DM;  // 1,048,576
  cvt_f32_bf16<<<WEL / 256, 256, 0, stream>>>(w_q, wbf + 0 * (size_t)WEL, WEL);
  cvt_f32_bf16<<<WEL / 256, 256, 0, stream>>>(w_k, wbf + 1 * (size_t)WEL, WEL);
  cvt_f32_bf16<<<WEL / 256, 256, 0, stream>>>(w_v, wbf + 2 * (size_t)WEL, WEL);
  cvt_f32_bf16<<<WEL / 256, 256, 0, stream>>>(w_o, wbf + 3 * (size_t)WEL, WEL);

  proj_qkv<<<dim3(BATCH * SEQL / 128, DM / 128, 3), 128, 0, stream>>>(
      query, key, value, wbf, b_q, b_k, b_v, qbf, kbf, vbf);

  vtrans<<<(int)(EL / 256), 256, 0, stream>>>(vbf, vtb);

  attn_kernel<<<BATCH * HEADS * (SEQL / 16) / 4, 128, 0, stream>>>(qbf, kbf, vtb, attn);

  out_proj<<<dim3(BATCH * SEQL / 128, DM / 128, 1), 128, 0, stream>>>(
      attn, wbf + 3 * (size_t)WEL, b_o, (float*)d_out);
}
